// MultiHeadAttention_7258494730307
// MI455X (gfx1250) — compile-verified, matrix operand order checked by reading, not run on hardware
//
#include <hip/hip_runtime.h>
#include <hip/hip_bf16.h>

// ---------------------------------------------------------------------------
// Causal multi-head attention forward for MI455X (gfx1250), wave32 + WMMA.
//   B=2, S=2048, D=1024, H=16, Dh=64.
// Pipeline:
//   1) qkv_kernel : QKV = x @ Wqkv^T   (f16 WMMA, f32 acc), scatter to
//                   Q[b,h,s,d] (pre-scaled 1/8), K[b,h,s,d], Vt[b,h,d,s] f16.
//   2) attn_kernel: flash attention per (b,h), online softmax, WMMA for both
//                   Q·K^T and P·V; P reshaped C-layout -> A-layout via LDS.
//   3) proj_kernel: out = attn_out @ Wo^T (f32 output).
// Workspace: 4 x 8MB f16 buffers = 32 MB.
// ---------------------------------------------------------------------------

typedef __attribute__((ext_vector_type(16))) _Float16 v16h;
typedef __attribute__((ext_vector_type(8)))  _Float16 v8h;
typedef __attribute__((ext_vector_type(8)))  float    v8f;
typedef __attribute__((ext_vector_type(4)))  float    v4f;

#define NB   2
#define NS   2048
#define ND   1024
#define NH   16
#define NDH  64

static __device__ inline v8f wmma16(v16h a, v16h b, v8f c) {
  // D = A(16x32 f16) * B(32x16 f16) + C(16x16 f32)
  return __builtin_amdgcn_wmma_f32_16x16x32_f16(
      /*neg_a=*/false, a, /*neg_b=*/false, b,
      /*c_mod=*/(short)0, c, /*reuse_a=*/false, /*reuse_b=*/false);
}

// Load one 16-lane-row A/B fragment from an f16 row-major matrix.
// p points at element [row][kbase + (lane>=16 ? 8 : 0)].
// ISA 16-bit A layout: elements 0..7 -> K+0..7, elements 8..15 -> K+16..23.
static __device__ inline v16h ldfrag_h(const _Float16* p) {
  v8h c0 = *(const v8h*)p;
  v8h c1 = *(const v8h*)(p + 16);
  v16h r;
#pragma unroll
  for (int j = 0; j < 8; ++j) { r[j] = c0[j]; r[j + 8] = c1[j]; }
  return r;
}

// Same fragment, but source matrix is f32 (convert on the fly).
static __device__ inline v16h ldfrag_f(const float* p) {
  v4f a0 = *(const v4f*)p;
  v4f a1 = *(const v4f*)(p + 4);
  v4f b0 = *(const v4f*)(p + 16);
  v4f b1 = *(const v4f*)(p + 20);
  v16h r;
#pragma unroll
  for (int j = 0; j < 4; ++j) {
    r[j]      = (_Float16)a0[j];
    r[j + 4]  = (_Float16)a1[j];
    r[j + 8]  = (_Float16)b0[j];
    r[j + 12] = (_Float16)b1[j];
  }
  return r;
}

// Row reductions across one 16-lane half of a wave32 (xor masks 1..8 stay
// inside each half) -- matches WMMA C-fragment geometry.
static __device__ inline float redmax16(float v) {
#pragma unroll
  for (int m = 1; m < 16; m <<= 1) v = fmaxf(v, __shfl_xor(v, m, 32));
  return v;
}
static __device__ inline float redadd16(float v) {
#pragma unroll
  for (int m = 1; m < 16; m <<= 1) v += __shfl_xor(v, m, 32);
  return v;
}

// ---------------------------------------------------------------------------
// Kernel 1: QKV projection.  M=4096 (b*s), N=3072, K=1024.
// Block = 256 threads = 8 waves (2x4), wave tile 64x64, block tile 128x256.
// ---------------------------------------------------------------------------
__global__ __launch_bounds__(256) void qkv_kernel(
    const float* __restrict__ x, const float* __restrict__ w_qkv,
    _Float16* __restrict__ Qh, _Float16* __restrict__ Kh,
    _Float16* __restrict__ Vt) {
  const int lane   = threadIdx.x & 31;
  const int wav    = threadIdx.x >> 5;
  const int half16 = lane >> 4;          // lane group (0: lanes 0-15)
  const int l16    = lane & 15;
  const int hoff   = half16 * 8;         // K offset of this lane group

  const int m0 = blockIdx.y * 128 + (wav >> 2) * 64;
  const int n0 = blockIdx.x * 256 + (wav & 3) * 64;

  v8f acc[4][4];
#pragma unroll
  for (int r = 0; r < 4; ++r)
#pragma unroll
    for (int c = 0; c < 4; ++c)
#pragma unroll
      for (int j = 0; j < 8; ++j) acc[r][c][j] = 0.0f;

  for (int k0 = 0; k0 < ND; k0 += 32) {
    v16h af[4], bf[4];
#pragma unroll
    for (int r = 0; r < 4; ++r)
      af[r] = ldfrag_f(x + (size_t)(m0 + r * 16 + l16) * ND + k0 + hoff);
#pragma unroll
    for (int c = 0; c < 4; ++c)
      bf[c] = ldfrag_f(w_qkv + (size_t)(n0 + c * 16 + l16) * ND + k0 + hoff);
#pragma unroll
    for (int r = 0; r < 4; ++r)
#pragma unroll
      for (int c = 0; c < 4; ++c) acc[r][c] = wmma16(af[r], bf[c], acc[r][c]);
  }

  // Scatter: the 64-wide wave tile sits inside exactly one {part, head}.
  const int part = n0 >> 10;             // 0=Q 1=K 2=V
  const int head = (n0 & 1023) >> 6;
#pragma unroll
  for (int c = 0; c < 4; ++c) {
#pragma unroll
    for (int r = 0; r < 4; ++r) {
#pragma unroll
      for (int vr = 0; vr < 8; ++vr) {
        const int m = m0 + r * 16 + vr + half16 * 8;   // C-layout row
        const int b = m >> 11, s = m & 2047;
        const int d = c * 16 + l16;                    // C-layout col
        const float v = acc[r][c][vr];
        if (part == 0) {
          Qh[(((b * NH + head) * NS) + s) * NDH + d] = (_Float16)(v * 0.125f);
        } else if (part == 1) {
          Kh[(((b * NH + head) * NS) + s) * NDH + d] = (_Float16)v;
        } else {
          Vt[(((b * NH + head) * NDH) + d) * NS + s] = (_Float16)v;  // V^T
        }
      }
    }
  }
}

// ---------------------------------------------------------------------------
// Kernel 2: causal flash attention.  Wave tile = 32 query rows; key tiles of
// 32 streamed up to the diagonal.  Q is pre-scaled by 1/sqrt(Dh).
// ---------------------------------------------------------------------------
__global__ __launch_bounds__(256) void attn_kernel(
    const _Float16* __restrict__ Qh, const _Float16* __restrict__ Kh,
    const _Float16* __restrict__ Vt, _Float16* __restrict__ AOh) {
  __shared__ __align__(16) _Float16 Plds[8][32][32];   // per-wave P staging

  const int lane   = threadIdx.x & 31;
  const int wav    = threadIdx.x >> 5;
  const int half16 = lane >> 4;
  const int l16    = lane & 15;
  const int hoff   = half16 * 8;

  const int b  = blockIdx.z;
  const int h  = blockIdx.y;
  const int q0 = blockIdx.x * 256 + wav * 32;          // 32 query rows / wave

  const _Float16* Qb = Qh + (size_t)((b * NH + h) * NS) * NDH;
  const _Float16* Kb = Kh + (size_t)((b * NH + h) * NS) * NDH;
  const _Float16* Vb = Vt + (size_t)((b * NH + h) * NDH) * NS;

  // Resident Q fragments: 2 row-tiles x 2 K-steps over Dh=64.
  v16h qf[2][2];
#pragma unroll
  for (int r = 0; r < 2; ++r)
#pragma unroll
    for (int kk = 0; kk < 2; ++kk)
      qf[r][kk] = ldfrag_h(Qb + (size_t)(q0 + r * 16 + l16) * NDH + kk * 32 + hoff);

  v8f Of[2][4];
  float mrow[2][8], lrow[2][8];
#pragma unroll
  for (int r = 0; r < 2; ++r) {
#pragma unroll
    for (int n = 0; n < 4; ++n)
#pragma unroll
      for (int j = 0; j < 8; ++j) Of[r][n][j] = 0.0f;
#pragma unroll
    for (int j = 0; j < 8; ++j) { mrow[r][j] = -1e30f; lrow[r][j] = 0.0f; }
  }

  const int nkt = (q0 >> 5) + 1;                       // causal tile count
  for (int kt = 0; kt < nkt; ++kt) {
    const int kb = kt * 32;

    // --- S = Q K^T for 32 keys (f32, scale already folded into Q) ---
    v8f sf[2][2];
#pragma unroll
    for (int r = 0; r < 2; ++r)
#pragma unroll
      for (int c = 0; c < 2; ++c)
#pragma unroll
        for (int j = 0; j < 8; ++j) sf[r][c][j] = 0.0f;

#pragma unroll
    for (int c = 0; c < 2; ++c)
#pragma unroll
      for (int kk = 0; kk < 2; ++kk) {
        v16h kf = ldfrag_h(Kb + (size_t)(kb + c * 16 + l16) * NDH + kk * 32 + hoff);
#pragma unroll
        for (int r = 0; r < 2; ++r) sf[r][c] = wmma16(qf[r][kk], kf, sf[r][c]);
      }

    // --- causal mask: only the diagonal tile needs it (kb == q0) ---
    if (kb == q0) {
#pragma unroll
      for (int r = 0; r < 2; ++r)
#pragma unroll
        for (int c = 0; c < 2; ++c)
#pragma unroll
          for (int vr = 0; vr < 8; ++vr) {
            const int qr = r * 16 + vr + half16 * 8;   // row within tile
            const int kc = c * 16 + l16;               // key within tile
            if (kc > qr) sf[r][c][vr] = -1e30f;
          }
    }

    // --- online softmax update (per C-fragment row) ---
#pragma unroll
    for (int r = 0; r < 2; ++r) {
#pragma unroll
      for (int vr = 0; vr < 8; ++vr) {
        float s0 = sf[r][0][vr], s1 = sf[r][1][vr];
        float tmax = redmax16(fmaxf(s0, s1));
        const float mold = mrow[r][vr];
        const float mnew = fmaxf(mold, tmax);
        const float alpha = __expf(mold - mnew);
        const float p0 = __expf(s0 - mnew);
        const float p1 = __expf(s1 - mnew);
        const float rs = redadd16(p0 + p1);
        lrow[r][vr] = lrow[r][vr] * alpha + rs;
        mrow[r][vr] = mnew;
#pragma unroll
        for (int n = 0; n < 4; ++n) Of[r][n][vr] *= alpha;
        sf[r][0][vr] = p0;
        sf[r][1][vr] = p1;
      }
    }

    // --- reshape P: C-layout -> A-layout via per-wave LDS slab ---
#pragma unroll
    for (int r = 0; r < 2; ++r)
#pragma unroll
      for (int c = 0; c < 2; ++c)
#pragma unroll
        for (int vr = 0; vr < 8; ++vr)
          Plds[wav][r * 16 + vr + half16 * 8][c * 16 + l16] =
              (_Float16)sf[r][c][vr];
    asm volatile("s_wait_dscnt 0" ::: "memory");

    v16h pf[2];
#pragma unroll
    for (int r = 0; r < 2; ++r)
      pf[r] = ldfrag_h(&Plds[wav][r * 16 + l16][hoff]);

    // --- O += P * V (V^T layout makes B-fragments contiguous) ---
#pragma unroll
    for (int n = 0; n < 4; ++n) {
      v16h vf = ldfrag_h(Vb + (size_t)(n * 16 + l16) * NS + kb + hoff);
#pragma unroll
      for (int r = 0; r < 2; ++r) Of[r][n] = wmma16(pf[r], vf, Of[r][n]);
    }
  }

  // --- finalize: divide by row sums, store f16 attn-out [b*s, h*64+d] ---
#pragma unroll
  for (int r = 0; r < 2; ++r)
#pragma unroll
    for (int vr = 0; vr < 8; ++vr) {
      const float inv = 1.0f / lrow[r][vr];
      const int s = q0 + r * 16 + vr + half16 * 8;
#pragma unroll
      for (int n = 0; n < 4; ++n)
        AOh[(size_t)(b * NS + s) * ND + h * NDH + n * 16 + l16] =
            (_Float16)(Of[r][n][vr] * inv);
    }
}

// ---------------------------------------------------------------------------
// Kernel 3: output projection.  out[4096,1024] f32 = AO(f16) @ Wo^T(f32->f16).
// ---------------------------------------------------------------------------
__global__ __launch_bounds__(256) void proj_kernel(
    const _Float16* __restrict__ AOh, const float* __restrict__ w_o,
    float* __restrict__ out) {
  const int lane   = threadIdx.x & 31;
  const int wav    = threadIdx.x >> 5;
  const int half16 = lane >> 4;
  const int l16    = lane & 15;
  const int hoff   = half16 * 8;

  const int m0 = blockIdx.y * 128 + (wav >> 2) * 64;
  const int n0 = blockIdx.x * 256 + (wav & 3) * 64;

  v8f acc[4][4];
#pragma unroll
  for (int r = 0; r < 4; ++r)
#pragma unroll
    for (int c = 0; c < 4; ++c)
#pragma unroll
      for (int j = 0; j < 8; ++j) acc[r][c][j] = 0.0f;

  for (int k0 = 0; k0 < ND; k0 += 32) {
    v16h af[4], bf[4];
#pragma unroll
    for (int r = 0; r < 4; ++r)
      af[r] = ldfrag_h(AOh + (size_t)(m0 + r * 16 + l16) * ND + k0 + hoff);
#pragma unroll
    for (int c = 0; c < 4; ++c)
      bf[c] = ldfrag_f(w_o + (size_t)(n0 + c * 16 + l16) * ND + k0 + hoff);
#pragma unroll
    for (int r = 0; r < 4; ++r)
#pragma unroll
      for (int c = 0; c < 4; ++c) acc[r][c] = wmma16(af[r], bf[c], acc[r][c]);
  }

#pragma unroll
  for (int r = 0; r < 4; ++r)
#pragma unroll
    for (int c = 0; c < 4; ++c)
#pragma unroll
      for (int vr = 0; vr < 8; ++vr) {
        const int m = m0 + r * 16 + vr + half16 * 8;
        const int n = n0 + c * 16 + l16;
        out[(size_t)m * ND + n] = acc[r][c][vr];
      }
}

// ---------------------------------------------------------------------------
// Host launcher.
// ---------------------------------------------------------------------------
extern "C" void kernel_launch(void* const* d_in, const int* in_sizes, int n_in,
                              void* d_out, int out_size, void* d_ws,
                              size_t ws_size, hipStream_t stream) {
  (void)in_sizes; (void)n_in; (void)out_size; (void)ws_size;

  const float* x     = (const float*)d_in[0];   // [2, 2048, 1024]
  const float* w_qkv = (const float*)d_in[1];   // [3072, 1024]
  const float* w_o   = (const float*)d_in[2];   // [1024, 1024]
  float*       out   = (float*)d_out;           // [2, 2048, 1024]

  // Workspace layout (f16): 4 buffers x 4M elements = 32 MB total.
  _Float16* ws  = (_Float16*)d_ws;
  const size_t SEG = (size_t)NB * NH * NS * NDH;   // 4,194,304
  _Float16* Qh  = ws;            // Q  [b,h,s,d], pre-scaled by 1/8
  _Float16* Kh  = ws + SEG;      // K  [b,h,s,d]
  _Float16* Vt  = ws + 2 * SEG;  // V^T[b,h,d,s]
  _Float16* AOh = ws + 3 * SEG;  // attn-out [b*s, h*64+d]

  qkv_kernel<<<dim3(3072 / 256, 4096 / 128), 256, 0, stream>>>(x, w_qkv, Qh, Kh, Vt);
  attn_kernel<<<dim3(NS / 256, NH, NB), 256, 0, stream>>>(Qh, Kh, Vt, AOh);
  proj_kernel<<<dim3(1024 / 256, 4096 / 128), 256, 0, stream>>>(AOh, w_o, out);
}
